// PQCModel_42314017800291
// MI455X (gfx1250) — compile-verified, matrix operand order checked by reading, not run on hardware
//
#include <hip/hip_runtime.h>

// ---------------------------------------------------------------------------
// Fused PQC kernel for MI455X (gfx1250, wave32):
//   h   = tanh(x @ W_projT) * scale            (WMMA f32_16x16x32_f16)
//   phi = h + weights[0]                       (adjacent RYs compose additively)
//   out = <Z_i> of CNOTchain(RY(w1)(CNOTchain(product(phi))))
// Roofline: dominant traffic = one read of x (201 MB) -> ~8.6 us @ 23.3 TB/s.
// Single fused pass, no intermediate HBM traffic; sim phase kept below the
// memory-bound budget via register/cross-lane gates (no LDS state).
// ---------------------------------------------------------------------------

typedef __attribute__((ext_vector_type(16))) _Float16 v16h;
typedef __attribute__((ext_vector_type(8)))  float    v8f;

#define SBERT   768
#define KCHUNKS (SBERT / 32)

// Convert 8 consecutive f32 -> f16 into dst[base..base+7] (base is 0 or 8).
__device__ __forceinline__ void cvt8(const float* __restrict__ p, v16h& d, int base) {
  float4 u = *(const float4*)(p);
  float4 w = *(const float4*)(p + 4);
  d[base + 0] = (_Float16)u.x; d[base + 1] = (_Float16)u.y;
  d[base + 2] = (_Float16)u.z; d[base + 3] = (_Float16)u.w;
  d[base + 4] = (_Float16)w.x; d[base + 5] = (_Float16)w.y;
  d[base + 6] = (_Float16)w.z; d[base + 7] = (_Float16)w.w;
}

// Full CNOT chain q=0..6 (ctrl=q, tgt=q+1) as ONE fused permutation.
// Basis map: |b> -> |prefixXOR(b)>, so out[y] = in[b(y)], b_p = y_p ^ y_{p+1}.
// Layout a = lane*8 + r (qubit q <-> bit 7-q):
//   srcLane(target P) = P ^ (P>>1)
//   srcReg (target r, P) = gray(r) ^ (P_bit0 ? 4 : 0), and the consumer of
//   lane Q has bit0 == parity(Q)  ->  lane-local register pre-permute + one
//   dynamic-index shuffle (ds_bpermute) per register.
__device__ __forceinline__ void cnot_chain(float v[8], int lane) {
  const bool c = (__popc(lane) & 1) != 0;   // consumer's low bit
  float t[8];
  t[0] = c ? v[4] : v[0];
  t[1] = c ? v[5] : v[1];
  t[2] = c ? v[7] : v[3];
  t[3] = c ? v[6] : v[2];
  t[4] = c ? v[2] : v[6];
  t[5] = c ? v[3] : v[7];
  t[6] = c ? v[1] : v[5];
  t[7] = c ? v[0] : v[4];
  const int srcLane = lane ^ (lane >> 1);
  #pragma unroll
  for (int r = 0; r < 8; ++r)
    v[r] = __shfl(t[r], srcLane, 32);       // gather: Dst[lane] = t[srcLane]
}

// RY on a register-index qubit bit: in-register 2x2 rotation.
__device__ __forceinline__ void ry_reg(float v[8], float c, float s, int bit) {
  #pragma unroll
  for (int r = 0; r < 8; ++r) {
    if (r & bit) continue;
    float a0 = v[r], a1 = v[r | bit];
    v[r]       = c * a0 - s * a1;
    v[r | bit] = fmaf(s, a0, c * a1);
  }
}

__global__ __launch_bounds__(256) void pqc_fused(
    const float* __restrict__ x,      // [B, 768]
    const float* __restrict__ Wp,     // [8, 768]
    const float* __restrict__ scalep, // [1]
    const float* __restrict__ wts,    // [2, 8]
    float* __restrict__ out,          // [B, 8]
    int batch)
{
  __shared__ float phi[8][16][8];     // [wave][sample-in-tile][qubit]

  const int lane = threadIdx.x & 31;
  const int wv   = threadIdx.x >> 5;
  const int tile = blockIdx.x * 8 + wv;
  const int row0 = tile * 16;
  if (row0 >= batch) return;          // terminated waves don't block barriers

  const int   nCol  = lane & 15;               // A row / B column for this lane
  const int   koff  = (lane < 16) ? 0 : 8;     // K half per 16-bit WMMA layout
  const float scale = scalep[0];

  const float* xrow = x  + (size_t)(row0 + nCol) * SBERT;
  const float* wrow = Wp + (size_t)(nCol & 7)    * SBERT; // clamp: cols 8..15 unused

  // ---------------- Phase 1: projection GEMM via WMMA ----------------
  v8f acc = {};
  #pragma unroll 4
  for (int kc = 0; kc < KCHUNKS; ++kc) {
    const int kb = kc * 32 + koff;
    // Speculative prefetch 4 chunks ahead (dropped silently past the buffer).
    __builtin_prefetch(xrow + kb + 128, 0, 1);
    v16h A, B;
    cvt8(xrow + kb,      A, 0);
    cvt8(xrow + kb + 16, A, 8);
    cvt8(wrow + kb,      B, 0);
    cvt8(wrow + kb + 16, B, 8);
    // (neg_a, A, neg_b, B, c_mod, C, reuse_a, reuse_b)
    acc = __builtin_amdgcn_wmma_f32_16x16x32_f16(
        false, A, false, B, (short)0, acc, false, false);
  }

  // phi[m][q] = tanh(h)*scale + w0[q]  (data-RY and layer-0 RY fuse additively)
  if (nCol < 8) {
    const float w0n   = wts[nCol];
    const int   mBase = (lane < 16) ? 0 : 8;     // C layout: VGPR r -> M = r (+8)
    #pragma unroll
    for (int r = 0; r < 8; ++r)
      phi[wv][mBase + r][nCol] = tanhf(acc[r]) * scale + w0n;
  }
  __syncthreads();

  // Layer-1 weight rotations (uniform across samples): cos/sin of w/2
  float cw[8], sw[8];
  #pragma unroll
  for (int q = 0; q < 8; ++q) {
    const float a = 0.5f * wts[8 + q];
    cw[q] = __cosf(a); sw[q] = __sinf(a);
  }

  // ---------------- Phase 2: statevector sim, one wave per sample ----------
  for (int s = 0; s < 16; ++s) {
    float cq[8], sq[8];
    #pragma unroll
    for (int q = 0; q < 8; ++q) {
      const float a = 0.5f * phi[wv][s][q];      // LDS broadcast read
      cq[q] = __cosf(a); sq[q] = __sinf(a);
    }

    // Product state: amp(a) = prod_q (bit_q ? sin : cos)(phi_q/2)
    float fl = ((lane & 16) ? sq[0] : cq[0]);
    fl      *= ((lane &  8) ? sq[1] : cq[1]);
    fl      *= ((lane &  4) ? sq[2] : cq[2]);
    fl      *= ((lane &  2) ? sq[3] : cq[3]);
    fl      *= ((lane &  1) ? sq[4] : cq[4]);
    float v[8];
    #pragma unroll
    for (int r = 0; r < 8; ++r) {
      const float g = ((r & 4) ? sq[5] : cq[5]) *
                      ((r & 2) ? sq[6] : cq[6]) *
                      ((r & 1) ? sq[7] : cq[7]);
      v[r] = fl * g;
    }

    cnot_chain(v, lane);

    // RY(w1) on all 8 qubits: qubits 0..4 via lane shuffles, 5..7 in-register
    #pragma unroll
    for (int q = 0; q < 5; ++q) {
      const int   mask = 1 << (4 - q);
      const bool  hi   = (lane & mask) != 0;
      const float ss   = hi ? sw[q] : -sw[q];
      const float cc   = cw[q];
      #pragma unroll
      for (int r = 0; r < 8; ++r) {
        float p = __shfl_xor(v[r], mask, 32);
        v[r] = fmaf(ss, p, cc * v[r]);
      }
    }
    ry_reg(v, cw[5], sw[5], 4);
    ry_reg(v, cw[6], sw[6], 2);
    ry_reg(v, cw[7], sw[7], 1);

    cnot_chain(v, lane);

    // <Z_i> = sum_a sign_i(a) * amp(a)^2
    float p2[8];
    #pragma unroll
    for (int r = 0; r < 8; ++r) p2[r] = v[r] * v[r];
    float sumAll = 0.f, z5 = 0.f, z6 = 0.f, z7 = 0.f;
    #pragma unroll
    for (int r = 0; r < 8; ++r) {
      sumAll += p2[r];
      z5 += (r & 4) ? -p2[r] : p2[r];
      z6 += (r & 2) ? -p2[r] : p2[r];
      z7 += (r & 1) ? -p2[r] : p2[r];
    }
    float z[8];
    z[0] = (lane & 16) ? -sumAll : sumAll;
    z[1] = (lane &  8) ? -sumAll : sumAll;
    z[2] = (lane &  4) ? -sumAll : sumAll;
    z[3] = (lane &  2) ? -sumAll : sumAll;
    z[4] = (lane &  1) ? -sumAll : sumAll;
    z[5] = z5; z[6] = z6; z[7] = z7;

    #pragma unroll
    for (int i = 0; i < 8; ++i) {
      z[i] += __shfl_xor(z[i], 16, 32);
      z[i] += __shfl_xor(z[i],  8, 32);
      z[i] += __shfl_xor(z[i],  4, 32);
      z[i] += __shfl_xor(z[i],  2, 32);
      z[i] += __shfl_xor(z[i],  1, 32);
    }
    if (lane == 0) {
      float4* dst = (float4*)(out + (size_t)(row0 + s) * 8);
      dst[0] = make_float4(z[0], z[1], z[2], z[3]);
      dst[1] = make_float4(z[4], z[5], z[6], z[7]);
    }
  }
}

extern "C" void kernel_launch(void* const* d_in, const int* in_sizes, int n_in,
                              void* d_out, int out_size, void* d_ws, size_t ws_size,
                              hipStream_t stream) {
  const float* x      = (const float*)d_in[0];
  const float* Wp     = (const float*)d_in[1];
  const float* scalep = (const float*)d_in[2];
  const float* wts    = (const float*)d_in[3];
  float*       out    = (float*)d_out;

  const int batch  = in_sizes[0] / SBERT;          // 65536
  const int tiles  = (batch + 15) / 16;            // 4096
  const int blocks = (tiles + 7) / 8;              // 512 blocks x 8 waves

  hipLaunchKernelGGL(pqc_fused, dim3(blocks), dim3(256), 0, stream,
                     x, Wp, scalep, wts, out, batch);
}
